// NPairLoss_26792005992921
// MI455X (gfx1250) — matrix operand order repaired and checked by reading, not compiled, not run on hardware
//
#include <hip/hip_runtime.h>
#include <math.h>

#define TWO_N 8192
#define NROWS 4096
#define DDIM  256
#define BK    64    // K-slice staged in LDS per phase (4 phases, double-buffered)
#define NPH   (DDIM / BK)
#define BSTR  72    // LDS row stride in bf16 (64 + 8 pad -> conflict-free banks)

typedef __attribute__((ext_vector_type(16))) __bf16 v16bf;
typedef __attribute__((ext_vector_type(8)))  __bf16 v8bf;
typedef __attribute__((ext_vector_type(4)))  __bf16 v4bf;
typedef __attribute__((ext_vector_type(8)))  float  v8f;
typedef __attribute__((ext_vector_type(4)))  float  v4f;
typedef __attribute__((ext_vector_type(4)))  int    v4i;

#if __has_builtin(__builtin_amdgcn_global_load_async_to_lds_b128)
#define USE_ASYNC_LDS 1
#else
#define USE_ASYNC_LDS 0
#endif

// ---------------------------------------------------------------------------
// Pass 1: split fp32 embeddings into bf16 hi/lo (x ~= hi + lo, lo = residual)
// ---------------------------------------------------------------------------
__global__ void split_bf16_kernel(const float* __restrict__ emb,
                                  __bf16* __restrict__ hi,
                                  __bf16* __restrict__ lo) {
  int idx = (blockIdx.x * blockDim.x + threadIdx.x) * 4;
  v4f x = *(const v4f*)(emb + idx);
  v4bf h, l;
#pragma unroll
  for (int t = 0; t < 4; ++t) {
    __bf16 hv = (__bf16)x[t];
    h[t] = hv;
    l[t] = (__bf16)(x[t] - (float)hv);
  }
  *(v4bf*)(hi + idx) = h;
  *(v4bf*)(lo + idx) = l;
}

// ---------------------------------------------------------------------------
// Pass 2: ap[i] = dot(E[i], E[i+N]) in full fp32; also zero row_sum[i].
// ---------------------------------------------------------------------------
__global__ void ap_kernel(const float* __restrict__ emb,
                          float* __restrict__ ap,
                          float* __restrict__ row_sum) {
  int lane = threadIdx.x & 31;
  int i    = blockIdx.x * (blockDim.x >> 5) + (threadIdx.x >> 5);
  const float* ei = emb + (size_t)i * DDIM;
  const float* ej = emb + (size_t)(i + NROWS) * DDIM;
  int k0 = lane * 8;
  v4f a0 = *(const v4f*)(ei + k0);
  v4f a1 = *(const v4f*)(ei + k0 + 4);
  v4f b0 = *(const v4f*)(ej + k0);
  v4f b1 = *(const v4f*)(ej + k0 + 4);
  float s = 0.f;
#pragma unroll
  for (int t = 0; t < 4; ++t) s = fmaf(a0[t], b0[t], s);
#pragma unroll
  for (int t = 0; t < 4; ++t) s = fmaf(a1[t], b1[t], s);
#pragma unroll
  for (int off = 16; off >= 1; off >>= 1) s += __shfl_xor(s, off, 32);
  if (lane == 0) { ap[i] = s; row_sum[i] = 0.f; }
}

// ---------------------------------------------------------------------------
// Pass 3: fused GEMM (bf16 WMMA, hi/lo compensated) + mask*exp + row reduce.
// Wave tile: 16 (M) x 64 (N). Block: 8 waves stacked in M -> 128 x 64.
// B operands double-buffered in LDS via async global->LDS copies (ASYNCcnt):
// phase p+1's copies are in flight while phase p feeds the WMMA pipe.
// ---------------------------------------------------------------------------
__device__ __forceinline__ v16bf cat8(v8bf a, v8bf b) {
  return __builtin_shufflevector(a, b, 0,1,2,3,4,5,6,7,8,9,10,11,12,13,14,15);
}

__device__ __forceinline__ void copy16_to_lds(const __bf16* g, __bf16* l) {
#if USE_ASYNC_LDS
  __builtin_amdgcn_global_load_async_to_lds_b128(
      (__attribute__((address_space(1))) v4i*)g,
      (__attribute__((address_space(3))) v4i*)l, 0, 0);
#else
  *(v8bf*)l = *(const v8bf*)g;
#endif
}

__device__ __forceinline__ void stage_phase(const __bf16* hi, const __bf16* lo,
                                            int j0, int p, int tid,
                                            __bf16 (*sBh)[BSTR],
                                            __bf16 (*sBl)[BSTR]) {
  // 512 16-byte chunks per array (64 cols x 8 chunks), 256 threads -> 2 each.
#pragma unroll
  for (int u = 0; u < 2; ++u) {
    int chunk = tid + u * 256;
    int col   = chunk >> 3;            // 0..63
    int kc    = (chunk & 7) << 3;      // 0,8,...,56
    size_t goff = (size_t)(j0 + col) * DDIM + p * BK + kc;
    copy16_to_lds(hi + goff, &sBh[col][kc]);
    copy16_to_lds(lo + goff, &sBl[col][kc]);
  }
}

__global__ void __launch_bounds__(256)
npair_gemm_kernel(const __bf16* __restrict__ hi,
                  const __bf16* __restrict__ lo,
                  const unsigned char* __restrict__ neg_mask,
                  const float* __restrict__ ap,
                  float* __restrict__ row_sum) {
  __shared__ __bf16 sBh[2][64][BSTR];
  __shared__ __bf16 sBl[2][64][BSTR];

  const int tid  = threadIdx.x;
  const int lane = tid & 31;
  const int wave = tid >> 5;
  const int mrow = lane & 15;          // A row-in-tile / B,C column-in-16
  const bool hiH = lane >= 16;
  const int i0   = blockIdx.y * 128 + wave * 16;
  const int j0   = blockIdx.x * 64;

  // Warm caches with this block's neg_mask tile (128 rows x 64 bytes).
  if (tid < 128)
    __builtin_prefetch(neg_mask + (size_t)(blockIdx.y * 128 + tid) * TWO_N + j0,
                       0, 1);

  // A-frag: lane holds row M=mrow, 16 contiguous K; lanes16-31 at K+16.
  const int aOff = hiH ? 16 : 0;
  // B-frag: lane holds column N=mrow, 16 contiguous K; lanes16-31 at K+16.
  const int bOff = hiH ? 16 : 0;

  const __bf16* aHiRow = hi + (size_t)(i0 + mrow) * DDIM;
  const __bf16* aLoRow = lo + (size_t)(i0 + mrow) * DDIM;

  v8f acc[4] = {v8f{0.f,0.f,0.f,0.f,0.f,0.f,0.f,0.f},
                v8f{0.f,0.f,0.f,0.f,0.f,0.f,0.f,0.f},
                v8f{0.f,0.f,0.f,0.f,0.f,0.f,0.f,0.f},
                v8f{0.f,0.f,0.f,0.f,0.f,0.f,0.f,0.f}};

  // Prologue: kick off phase 0's async copies.
  stage_phase(hi, lo, j0, 0, tid, sBh[0], sBl[0]);

#pragma unroll
  for (int p = 0; p < NPH; ++p) {
    const int buf = p & 1;
#if USE_ASYNC_LDS
    asm volatile("s_wait_asynccnt 0x0" ::: "memory");
#endif
    __syncthreads();   // phase-p data visible; buf^1 readers (phase p-1) done

    // Prefetch next phase into the other buffer while we compute on this one.
    if (p + 1 < NPH)
      stage_phase(hi, lo, j0, p + 1, tid, sBh[buf ^ 1], sBl[buf ^ 1]);

#pragma unroll
    for (int lk = 0; lk < BK; lk += 32) {
      const int kg = p * BK + lk;
      v16bf ah = cat8(*(const v8bf*)(aHiRow + kg + aOff),
                      *(const v8bf*)(aHiRow + kg + 8 + aOff));
      v16bf al = cat8(*(const v8bf*)(aLoRow + kg + aOff),
                      *(const v8bf*)(aLoRow + kg + 8 + aOff));
#pragma unroll
      for (int c = 0; c < 4; ++c) {
        const __bf16* bph = &sBh[buf][c * 16 + mrow][lk + bOff];
        const __bf16* bpl = &sBl[buf][c * 16 + mrow][lk + bOff];
        v16bf bh = cat8(*(const v8bf*)bph, *(const v8bf*)(bph + 8));
        v16bf bl = cat8(*(const v8bf*)bpl, *(const v8bf*)(bpl + 8));
        acc[c] = __builtin_amdgcn_wmma_f32_16x16x32_bf16(
            false, ah, false, bh, (short)0, acc[c], false, false);
        acc[c] = __builtin_amdgcn_wmma_f32_16x16x32_bf16(
            false, al, false, bh, (short)0, acc[c], false, false);
        acc[c] = __builtin_amdgcn_wmma_f32_16x16x32_bf16(
            false, ah, false, bl, (short)0, acc[c], false, false);
      }
    }
  }

  // Epilogue. C layout: VGPR r -> M = r (lanes 0-15) / r+8 (lanes 16-31),
  // N = (lane & 15). Fuse mask * exp(d - ap[i]) and reduce across the 16
  // lanes sharing a row, then one atomic per (row, block-tile).
  const int half = hiH ? 8 : 0;
#pragma unroll
  for (int r = 0; r < 8; ++r) {
    const int i = i0 + r + half;
    const float api = ap[i];
    float partial = 0.f;
#pragma unroll
    for (int c = 0; c < 4; ++c) {
      const int j = j0 + c * 16 + mrow;
      const float m = (float)neg_mask[(size_t)i * TWO_N + j];
      partial += m * __expf(acc[c][r] - api);
    }
#pragma unroll
    for (int off = 1; off < 16; off <<= 1)
      partial += __shfl_xor(partial, off, 16);
    if (mrow == 0) atomicAdd(&row_sum[i], partial);
  }
}

// ---------------------------------------------------------------------------
// Pass 4: out = mean(log1p(row_sum))
// ---------------------------------------------------------------------------
__global__ void finalize_kernel(const float* __restrict__ row_sum,
                                float* __restrict__ out) {
  __shared__ float red[256];
  float s = 0.f;
  for (int i = threadIdx.x; i < NROWS; i += 256) s += log1pf(row_sum[i]);
  red[threadIdx.x] = s;
  __syncthreads();
#pragma unroll
  for (int stride = 128; stride > 0; stride >>= 1) {
    if ((int)threadIdx.x < stride) red[threadIdx.x] += red[threadIdx.x + stride];
    __syncthreads();
  }
  if (threadIdx.x == 0) out[0] = red[0] / (float)NROWS;
}

// ---------------------------------------------------------------------------
extern "C" void kernel_launch(void* const* d_in, const int* in_sizes, int n_in,
                              void* d_out, int out_size, void* d_ws, size_t ws_size,
                              hipStream_t stream) {
  const float* emb              = (const float*)d_in[0];
  // d_in[1] = pos_mask: unused by the reference computation.
  const unsigned char* neg_mask = (const unsigned char*)d_in[2];
  float* out                    = (float*)d_out;

  // Workspace layout: ap[4096] | row_sum[4096] | hi[8192*256] | lo[8192*256]
  float*  ap      = (float*)d_ws;
  float*  row_sum = ap + NROWS;
  __bf16* hi      = (__bf16*)(row_sum + NROWS);
  __bf16* lo      = hi + (size_t)TWO_N * DDIM;

  split_bf16_kernel<<<(TWO_N * DDIM) / (256 * 4), 256, 0, stream>>>(emb, hi, lo);
  ap_kernel<<<NROWS / 8, 256, 0, stream>>>(emb, ap, row_sum);
  npair_gemm_kernel<<<dim3(TWO_N / 64, NROWS / 128), 256, 0, stream>>>(
      hi, lo, neg_mask, ap, row_sum);
  finalize_kernel<<<1, 256, 0, stream>>>(row_sum, out);
}
